// RegionSeparatedAttention_36086315221025
// MI455X (gfx1250) — matrix operand order repaired and checked by reading, compiled, not run on hardware
//
#include <hip/hip_runtime.h>
#include <hip/hip_bf16.h>

// RegionSeparatedAttention for MI455X (gfx1250, wave32, WMMA).
// Flash-style: never materialize the 512MB attention matrix; recompute score
// tiles with v_wmma_f32_16x16x32_bf16 in (1) a stats pass and (2) two output
// sweeps (A@P and B@P^T), staging normalized P tiles through LDS as bf16.

typedef __attribute__((ext_vector_type(16))) __bf16       bf16x16;
typedef __attribute__((ext_vector_type(2)))  __bf16       bf16x2;
typedef __attribute__((ext_vector_type(8)))  float        v8f;
typedef __attribute__((ext_vector_type(8)))  unsigned int v8u;

#define C64 64
#define HW  16384   // 128*128 pixels per batch image
#define NN  4096    // 64*64 pixels per region
#define NBR 8       // B(2) * R(4) region slots

// ---------- bf16 helpers: native converts (v_cvt_*bf16*) ----------
__device__ __forceinline__ unsigned pack2(float lo, float hi) {
  bf16x2 v = { (__bf16)lo, (__bf16)hi };
  return __builtin_bit_cast(unsigned, v);
}
__device__ __forceinline__ __bf16 bf1(float f) { return (__bf16)f; }

__device__ __forceinline__ float rcp_fast(float x) {
#if __has_builtin(__builtin_amdgcn_rcpf)
  return __builtin_amdgcn_rcpf(x);   // v_rcp_f32 (TRANS, co-executes with WMMA)
#else
  return 1.0f / x;
#endif
}

// ---------- WMMA operand gathers (wave32 lane layouts, ISA 7.12.2) ----------
// A operand 16x32 bf16: lane: M = lane&15, K(e) = 8*(lane>>4) + (e&7) + 16*(e>>3)
// (caller folds kb8 = 8*(lane>>4) and the per-lane row offset into args)
__device__ __forceinline__ bf16x16 gA16(const __bf16* src, int rowoff, int col0, int kb8) {
  const __bf16* p = src + rowoff + col0 + kb8;
  v8u u;
#pragma unroll
  for (int t = 0; t < 8; ++t) {
    int k = 2 * (t & 3) + 16 * (t >> 2);
    u[t] = *(const unsigned*)(p + k);
  }
  return __builtin_bit_cast(bf16x16, u);
}
// B operand 32x16 bf16: lane: N = lane&15, K(e) = 8*(lane>>4) + (e&7) + 16*(e>>3)
// (same order as gA16)
__device__ __forceinline__ bf16x16 gB16(const __bf16* src, int rowoff, int col0, int kb16) {
  const __bf16* p = src + rowoff + col0 + (kb16 >> 1);
  v8u u;
#pragma unroll
  for (int t = 0; t < 8; ++t) u[t] = *(const unsigned*)(p + 2 * (t & 3) + 16 * (t >> 2));
  return __builtin_bit_cast(bf16x16, u);
}
__device__ __forceinline__ v8f wmma_bf16(bf16x16 a, bf16x16 b, v8f c) {
  return __builtin_amdgcn_wmma_f32_16x16x32_bf16(false, a, false, b, (short)0, c,
                                                 false, false);
}

// =====================================================================
// Kernel 1: fused q/k/a/b 1x1-conv projections -> bf16 regionized layouts
//   qT,kT : [br][n][c]  (row = pixel, contiguous channels -> A/B gathers)
//   aM,bM : [br][c][n]  (row = channel, contiguous pixels)
// grid (2048 pixel-blocks, 4 matrices), block 128 = 4 waves (one per c-rowblock)
// =====================================================================
__global__ void __launch_bounds__(128) k_proj(
    const float* __restrict__ x,
    const float* __restrict__ W0, const float* __restrict__ B0,
    const float* __restrict__ W1, const float* __restrict__ B1,
    const float* __restrict__ W2, const float* __restrict__ B2,
    const float* __restrict__ W3, const float* __restrict__ B3,
    __bf16* __restrict__ qT, __bf16* __restrict__ kT,
    __bf16* __restrict__ aM, __bf16* __restrict__ bM) {
  const int lane = threadIdx.x & 31;
  const int wv   = threadIdx.x >> 5;
  const int m    = blockIdx.y;
  const float *W, *Bv;
  if (m == 0)      { W = W0; Bv = B0; }
  else if (m == 1) { W = W1; Bv = B1; }
  else if (m == 2) { W = W2; Bv = B2; }
  else             { W = W3; Bv = B3; }

  const int p0 = blockIdx.x * 16;
  const int b  = p0 >> 14;
  const int hw = p0 & (HW - 1);
  const int h  = hw >> 7, w0 = hw & 127;
  const int br = b * 4 + ((h >> 6) * 2 + (w0 >> 6));
  const int n0 = (h & 63) * 64 + (w0 & 63);

  const int hi = lane >> 4, lm = lane & 15;
  const int kb8 = hi * 8, kb16 = hi * 16;
  const int orow = wv * 16;

  v8f acc;                           // bias preloaded straight into C layout
#pragma unroll
  for (int j = 0; j < 8; ++j) acc[j] = Bv[orow + j + 8 * hi];

#pragma unroll
  for (int kk = 0; kk < 2; ++kk) {
    v8u ua;                          // A: W[orow+lm, c], fp32 -> bf16 pairs
    const float* wr = W + (orow + lm) * C64 + kk * 32 + kb8;
#pragma unroll
    for (int t = 0; t < 8; ++t) {
      int k = 2 * (t & 3) + 16 * (t >> 2);
      ua[t] = pack2(wr[k], wr[k + 1]);
    }
    v8u ub;                          // B: x[b, c, h, w0+lm], channel-strided
    const float* xp = x + (size_t)(b * C64 + kk * 32 + kb8) * HW + hw + lm;
#pragma unroll
    for (int t = 0; t < 8; ++t)
      ub[t] = pack2(xp[(size_t)(2 * (t & 3) + 16 * (t >> 2)) * HW], xp[(size_t)(2 * (t & 3) + 16 * (t >> 2) + 1) * HW]);
    acc = wmma_bf16(__builtin_bit_cast(bf16x16, ua),
                    __builtin_bit_cast(bf16x16, ub), acc);
  }

  if (m <= 1) {                      // transposed [n][c] store for q/k
    __bf16* dst = (m == 0 ? qT : kT) + ((size_t)br * NN + n0 + lm) * C64 + orow;
#pragma unroll
    for (int j = 0; j < 8; ++j) dst[j + 8 * hi] = bf1(acc[j]);
  } else {                           // [c][n] store for a/b
    __bf16* dst = (m == 2 ? aM : bM) + ((size_t)br * C64 + orow + 8 * hi) * NN + n0 + lm;
#pragma unroll
    for (int j = 0; j < 8; ++j) dst[(size_t)j * NN] = bf1(acc[j]);
  }
}

// =====================================================================
// Kernel 2: online-softmax statistics (row max M, row sum Z) per region row.
// grid (32, 8 br), block 256 = 8 waves, each wave owns 16 rows, sweeps 4096 cols.
// =====================================================================
__global__ void __launch_bounds__(256) k_stats(
    const __bf16* __restrict__ qT, const __bf16* __restrict__ kT,
    float2* __restrict__ MZ) {
  const int lane = threadIdx.x & 31;
  const int wv   = threadIdx.x >> 5;
  const int br   = blockIdx.y;
  const int n0   = (blockIdx.x * 8 + wv) * 16;
  const int hi = lane >> 4, lm = lane & 15;
  const int kb8 = hi * 8, kb16 = hi * 16;

  const __bf16* q = qT + (size_t)br * NN * C64;
  const __bf16* k = kT + (size_t)br * NN * C64;

  const bf16x16 A0 = gA16(q, (n0 + lm) * C64, 0, kb8);
  const bf16x16 A1 = gA16(q, (n0 + lm) * C64, 32, kb8);

  float runM[8], runZ[8];
#pragma unroll
  for (int j = 0; j < 8; ++j) { runM[j] = -1e30f; runZ[j] = 0.0f; }

  for (int m0 = 0; m0 < NN; m0 += 16) {
    bf16x16 Bb0 = gB16(k, (m0 + lm) * C64, 0, kb16);
    bf16x16 Bb1 = gB16(k, (m0 + lm) * C64, 32, kb16);
    v8f s = {};
    s = wmma_bf16(A0, Bb0, s);
    s = wmma_bf16(A1, Bb1, s);
#pragma unroll
    for (int j = 0; j < 8; ++j) {          // row spans 16 lanes of same half
      float v = s[j] * 0.125f;             // 1/sqrt(C)
      float tmax = v;
      tmax = fmaxf(tmax, __shfl_xor(tmax, 1, 32));
      tmax = fmaxf(tmax, __shfl_xor(tmax, 2, 32));
      tmax = fmaxf(tmax, __shfl_xor(tmax, 4, 32));
      tmax = fmaxf(tmax, __shfl_xor(tmax, 8, 32));
      float nm = fmaxf(runM[j], tmax);
      float p = __expf(v - nm);
      p += __shfl_xor(p, 1, 32);
      p += __shfl_xor(p, 2, 32);
      p += __shfl_xor(p, 4, 32);
      p += __shfl_xor(p, 8, 32);
      runZ[j] = runZ[j] * __expf(runM[j] - nm) + p;
      runM[j] = nm;
    }
  }
  if (lm == 0) {
#pragma unroll
    for (int j = 0; j < 8; ++j)
      MZ[(size_t)br * NN + n0 + j + 8 * hi] = make_float2(runM[j], runZ[j]);
  }
}

// =====================================================================
// Kernel 3: out1[:,m] = sum_j a[:,j] * P[j,m].  Group of 4 waves owns one
// 16-col m tile; per 128-row j chunk: recompute S via WMMA, normalize, stage
// bf16 P in padded LDS (D-layout -> B-operand layout), 4 chained out-WMMAs.
// grid (128, 8 br), block 256 = 2 groups x 4 waves.
// =====================================================================
__global__ void __launch_bounds__(256) k_out1(
    const __bf16* __restrict__ qT, const __bf16* __restrict__ kT,
    const __bf16* __restrict__ aM, const float2* __restrict__ MZ,
    float* __restrict__ outacc) {
  __shared__ __bf16 P[2][16][136];   // [group][m][j_local], padded vs conflicts
  const int lane = threadIdx.x & 31;
  const int wv = threadIdx.x >> 5;
  const int g = wv >> 2, w4 = wv & 3;
  const int br = blockIdx.y;
  const int m0 = (blockIdx.x * 2 + g) * 16;
  const int hi = lane >> 4, lm = lane & 15;
  const int kb8 = hi * 8, kb16 = hi * 16;
  const int c0 = w4 * 16;

  const __bf16* q = qT + (size_t)br * NN * C64;
  const __bf16* k = kT + (size_t)br * NN * C64;
  const __bf16* a = aM + (size_t)br * C64 * NN;
  const float2* mz = MZ + (size_t)br * NN;

  const bf16x16 Bk0 = gB16(k, (m0 + lm) * C64, 0, kb16);   // K cols fixed: hoist
  const bf16x16 Bk1 = gB16(k, (m0 + lm) * C64, 32, kb16);

  v8f oacc = {};
  for (int jc = 0; jc < NN; jc += 128) {
    const int jr0 = jc + w4 * 32;
#pragma unroll
    for (int t = 0; t < 2; ++t) {
      const int jrow = jr0 + t * 16;
      bf16x16 Aq0 = gA16(q, (jrow + lm) * C64, 0, kb8);
      bf16x16 Aq1 = gA16(q, (jrow + lm) * C64, 32, kb8);
      v8f s = {};
      s = wmma_bf16(Aq0, Bk0, s);
      s = wmma_bf16(Aq1, Bk1, s);
      float pv[8];
#pragma unroll
      for (int vg = 0; vg < 8; ++vg) {     // normalize per score-row j
        float2 z = mz[jrow + vg + 8 * hi];
        pv[vg] = __expf(s[vg] * 0.125f - z.x) * rcp_fast(z.y);
      }
      unsigned* prow = (unsigned*)&P[g][lm][w4 * 32 + t * 16 + 8 * hi];
#pragma unroll
      for (int qq = 0; qq < 4; ++qq) prow[qq] = pack2(pv[2 * qq], pv[2 * qq + 1]);
    }
    __syncthreads();
    const __bf16* ar = a + (size_t)(c0 + lm) * NN + jc;
    __builtin_prefetch(ar + 128, 0, 1);    // global_prefetch_b8 next chunk
#pragma unroll
    for (int kk = 0; kk < 4; ++kk) {
      bf16x16 Aa = gA16(ar, 0, kk * 32, kb8);
      bf16x16 Bp = gB16(&P[g][0][0], lm * 136, kk * 32, kb16);
      oacc = wmma_bf16(Aa, Bp, oacc);
    }
    __syncthreads();
  }
  float* dst = outacc + ((size_t)br * C64 + c0 + 8 * hi) * NN + m0 + lm;
#pragma unroll
  for (int vg = 0; vg < 8; ++vg) dst[(size_t)vg * NN] = oacc[vg];
}

// =====================================================================
// Kernel 4: out2[:,n] += sum_j b[:,j] * P[n,j].  Same structure; normalizer
// is per output column n (hoisted), P staged transposed into LDS.
// =====================================================================
__global__ void __launch_bounds__(256) k_out2(
    const __bf16* __restrict__ qT, const __bf16* __restrict__ kT,
    const __bf16* __restrict__ bM, const float2* __restrict__ MZ,
    float* __restrict__ outacc) {
  __shared__ __bf16 P[2][16][136];   // [group][n][j_local]
  const int lane = threadIdx.x & 31;
  const int wv = threadIdx.x >> 5;
  const int g = wv >> 2, w4 = wv & 3;
  const int br = blockIdx.y;
  const int n0 = (blockIdx.x * 2 + g) * 16;
  const int hi = lane >> 4, lm = lane & 15;
  const int kb8 = hi * 8, kb16 = hi * 16;
  const int c0 = w4 * 16;

  const __bf16* q = qT + (size_t)br * NN * C64;
  const __bf16* k = kT + (size_t)br * NN * C64;
  const __bf16* bm = bM + (size_t)br * C64 * NN;
  const float2* mz = MZ + (size_t)br * NN;

  const bf16x16 Aq0 = gA16(q, (n0 + lm) * C64, 0, kb8);    // Q rows fixed: hoist
  const bf16x16 Aq1 = gA16(q, (n0 + lm) * C64, 32, kb8);
  float rM[8], rR[8];
#pragma unroll
  for (int vg = 0; vg < 8; ++vg) {
    float2 z = mz[n0 + vg + 8 * hi];
    rM[vg] = z.x; rR[vg] = rcp_fast(z.y);
  }

  v8f oacc = {};
  for (int jc = 0; jc < NN; jc += 128) {
    const int jr0 = jc + w4 * 32;
#pragma unroll
    for (int t = 0; t < 2; ++t) {
      const int jcol = jr0 + t * 16;
      bf16x16 Bk0 = gB16(k, (jcol + lm) * C64, 0, kb16);
      bf16x16 Bk1 = gB16(k, (jcol + lm) * C64, 32, kb16);
      v8f s = {};
      s = wmma_bf16(Aq0, Bk0, s);
      s = wmma_bf16(Aq1, Bk1, s);
#pragma unroll
      for (int vg = 0; vg < 8; ++vg) {     // transpose-store normalized P
        float p = __expf(s[vg] * 0.125f - rM[vg]) * rR[vg];
        P[g][vg + 8 * hi][w4 * 32 + t * 16 + lm] = bf1(p);
      }
    }
    __syncthreads();
    const __bf16* brp = bm + (size_t)(c0 + lm) * NN + jc;
    __builtin_prefetch(brp + 128, 0, 1);
#pragma unroll
    for (int kk = 0; kk < 4; ++kk) {
      bf16x16 Ab = gA16(brp, 0, kk * 32, kb8);
      bf16x16 Bp = gB16(&P[g][0][0], lm * 136, kk * 32, kb16);
      oacc = wmma_bf16(Ab, Bp, oacc);
    }
    __syncthreads();
  }
  float* dst = outacc + ((size_t)br * C64 + c0 + 8 * hi) * NN + n0 + lm;
#pragma unroll
  for (int vg = 0; vg < 8; ++vg) dst[(size_t)vg * NN] += oacc[vg];
}

// =====================================================================
// Kernel 5: final 1x1 conv Wo @ out + bo, de-regionize on the output side.
// =====================================================================
__global__ void __launch_bounds__(128) k_final(
    const float* __restrict__ oa, const float* __restrict__ Wo,
    const float* __restrict__ bo, float* __restrict__ y) {
  const int lane = threadIdx.x & 31;
  const int wv   = threadIdx.x >> 5;
  const int p0 = blockIdx.x * 16;
  const int b  = p0 >> 14;
  const int hw = p0 & (HW - 1);
  const int h  = hw >> 7, w0 = hw & 127;
  const int br = b * 4 + ((h >> 6) * 2 + (w0 >> 6));
  const int n0 = (h & 63) * 64 + (w0 & 63);
  const int hi = lane >> 4, lm = lane & 15;
  const int kb8 = hi * 8, kb16 = hi * 16;
  const int orow = wv * 16;

  v8f acc;
#pragma unroll
  for (int j = 0; j < 8; ++j) acc[j] = bo[orow + j + 8 * hi];

#pragma unroll
  for (int kk = 0; kk < 2; ++kk) {
    v8u ua;
    const float* wr = Wo + (orow + lm) * C64 + kk * 32 + kb8;
#pragma unroll
    for (int t = 0; t < 8; ++t) {
      int k = 2 * (t & 3) + 16 * (t >> 2);
      ua[t] = pack2(wr[k], wr[k + 1]);
    }
    v8u ub;
    const float* src = oa + (size_t)(br * C64 + kk * 32 + kb8) * NN + n0 + lm;
#pragma unroll
    for (int t = 0; t < 8; ++t)
      ub[t] = pack2(src[(size_t)(2 * (t & 3) + 16 * (t >> 2)) * NN], src[(size_t)(2 * (t & 3) + 16 * (t >> 2) + 1) * NN]);
    acc = wmma_bf16(__builtin_bit_cast(bf16x16, ua),
                    __builtin_bit_cast(bf16x16, ub), acc);
  }
  float* dst = y + (size_t)(b * C64 + orow + 8 * hi) * HW + hw + lm;
#pragma unroll
  for (int j = 0; j < 8; ++j) dst[(size_t)j * HW] = acc[j];
}

// =====================================================================
extern "C" void kernel_launch(void* const* d_in, const int* in_sizes, int n_in,
                              void* d_out, int out_size, void* d_ws, size_t ws_size,
                              hipStream_t stream) {
  const float* x  = (const float*)d_in[0];
  const float* Wq = (const float*)d_in[1]; const float* bq = (const float*)d_in[2];
  const float* Wk = (const float*)d_in[3]; const float* bk = (const float*)d_in[4];
  const float* Wa = (const float*)d_in[5]; const float* ba = (const float*)d_in[6];
  const float* Wb = (const float*)d_in[7]; const float* bb = (const float*)d_in[8];
  const float* Wo = (const float*)d_in[9]; const float* bo = (const float*)d_in[10];
  float* y = (float*)d_out;

  // Workspace layout (~24.3 MB): 4x bf16 projections + stats + fp32 accumulator
  char* ws = (char*)d_ws;
  const size_t SZbf = (size_t)NBR * NN * C64 * sizeof(__bf16);  // 4 MB each
  __bf16* qT = (__bf16*)(ws);
  __bf16* kT = (__bf16*)(ws + SZbf);
  __bf16* aM = (__bf16*)(ws + 2 * SZbf);
  __bf16* bM = (__bf16*)(ws + 3 * SZbf);
  float2* MZ = (float2*)(ws + 4 * SZbf);                        // 256 KB
  float*  oa = (float*) (ws + 4 * SZbf + (size_t)NBR * NN * sizeof(float2)); // 8 MB

  k_proj <<<dim3(2048, 4), dim3(128), 0, stream>>>(x, Wq, bq, Wk, bk, Wa, ba,
                                                   Wb, bb, qT, kT, aM, bM);
  k_stats<<<dim3(32, 8),   dim3(256), 0, stream>>>(qT, kT, MZ);
  k_out1 <<<dim3(128, 8),  dim3(256), 0, stream>>>(qT, kT, aM, MZ, oa);
  k_out2 <<<dim3(128, 8),  dim3(256), 0, stream>>>(qT, kT, bM, MZ, oa);
  k_final<<<dim3(2048, 1), dim3(128), 0, stream>>>(oa, Wo, bo, y);
}
